// MiniDetector_63127429316777
// MI455X (gfx1250) — matrix operand order repaired and checked by reading, not compiled, not run on hardware
//
#include <hip/hip_runtime.h>
#include <math.h>

// ---------------------------------------------------------------------------
// Types for CDNA5 WMMA (wave32): v_wmma_f32_16x16x32_bf16
// ---------------------------------------------------------------------------
typedef __attribute__((ext_vector_type(16))) __bf16         bf16x16;
typedef __attribute__((ext_vector_type(8)))  float          f32x8;
typedef __attribute__((ext_vector_type(4)))  float          f32x4;
typedef __attribute__((ext_vector_type(16))) unsigned short u16x16;
typedef __attribute__((ext_vector_type(8)))  unsigned short u16x8;
typedef __attribute__((ext_vector_type(4)))  unsigned short u16x4;

union CastBF  { u16x16 u; bf16x16 b; };
union CastH8  { u16x8 h[2]; u16x16 v; };

__device__ __forceinline__ unsigned short f2bf(float f) {
  unsigned int u = __builtin_bit_cast(unsigned int, f);
  u += 0x7FFFu + ((u >> 16) & 1u);          // round-to-nearest-even
  return (unsigned short)(u >> 16);
}
__device__ __forceinline__ float bf2f(unsigned short h) {
  unsigned int u = ((unsigned int)h) << 16;
  return __builtin_bit_cast(float, u);
}
__device__ __forceinline__ float sigmoidf_(float x) { return 1.0f / (1.0f + __expf(-x)); }

__device__ __forceinline__ f32x8 zero8() {
  f32x8 z = {0.f,0.f,0.f,0.f,0.f,0.f,0.f,0.f}; return z;
}
__device__ __forceinline__ f32x8 wmma_bf16(bf16x16 a, bf16x16 b, f32x8 c) {
  return __builtin_amdgcn_wmma_f32_16x16x32_bf16(false, a, false, b, (short)0, c,
                                                 false, false);
}

// Problem constants
#define BATCH 1024
#define SEQ   49        // 7*7
#define HID   256
#define FEAT  12544     // 49*256
#define ABUF  12800     // FEAT + one always-zero row (256)
#define CLS   80
#define TOPK  32
#define ALLP  84        // CLS + 4

// sine-embed exponents: 10000^(-2*(j/2)/half) = exp2(-(j/2)*log2(1e4)/(half/2))
#define NEG_L2_1E4_D64  (-0.20762050593046f)   // log2(10000)/64  (half = 128)
#define NEG_L2_1E4_D128 (-0.10381025296523f)   // log2(10000)/128 (half = 256)
#define TWO_PI 6.2831853071795864f

// ---------------------------------------------------------------------------
// Pack a row-major K x N fp32 weight matrix into the WMMA B-fragment layout:
// packed[((t*Ntiles + nt)*32 + lane)*16 + i] = bf16( W[(t*32+(lane/16)*8+(i%8)+(i/8)*16)*N
//                                                     + nt*16 + lane%16] )
// ---------------------------------------------------------------------------
__global__ void packb_kernel(const float* __restrict__ W,
                             unsigned short* __restrict__ out,
                             int Ntiles, int N, int total) {
  int e = blockIdx.x * blockDim.x + threadIdx.x;
  if (e >= total) return;
  int i    = e & 15;
  int lane = (e >> 4) & 31;
  int tn   = e >> 9;                 // t*Ntiles + nt
  int nt   = tn % Ntiles;
  int t    = tn / Ntiles;
  int k    = t * 32 + ((lane >> 4) << 3) + (i & 7) + ((i >> 3) << 4);
  int n    = nt * 16 + (lane & 15);
  out[e] = (n < N) ? f2bf(W[(size_t)k * N + n]) : (unsigned short)0;
}

// vectorized f32 -> bf16 LDS fill (plus zero row at FEAT..FEAT+255)
__device__ __forceinline__ void fill_bf16(unsigned short* dst,
                                          const float* __restrict__ src, int tid) {
  for (int i = tid * 4; i < FEAT; i += 1024) {
    f32x4 v = *(const f32x4*)(src + i);
    u16x4 h = { f2bf(v.x), f2bf(v.y), f2bf(v.z), f2bf(v.w) };
    *(u16x4*)(dst + i) = h;
  }
  if (tid < 256) dst[FEAT + tid] = 0;
}

// Fused DAB-DETR sine embedding of the raw input (d_model=256) straight into
// the LDS activation buffer: cols 0..127 = emb(ch 1), 128..255 = emb(ch 0).
// Each 4-wide group shares one coordinate (groups never straddle col 128).
__device__ __forceinline__ void fill_sine(unsigned short* dst,
                                          const float* __restrict__ src, int tid) {
  for (int i = tid * 4; i < FEAT; i += 1024) {
    int c = i & 255;
    float coord = src[(i & ~255) + ((c < 128) ? 1 : 0)];
    u16x4 h;
#pragma unroll
    for (int q = 0; q < 4; ++q) {
      int j = (c + q) & 127;
      float freq = __builtin_exp2f((float)(j >> 1) * NEG_L2_1E4_D64);
      float arg  = coord * TWO_PI * freq;
      h[q] = f2bf((j & 1) ? __cosf(arg) : __sinf(arg));
    }
    *(u16x4*)(dst + i) = h;
  }
  if (tid < 256) dst[FEAT + tid] = 0;
}

// ---------------------------------------------------------------------------
// Conv stack as implicit GEMM: one workgroup (8 wave32) per image.
// Activations ping-pong in LDS as bf16 (49x256 + zero row each).
// Outer loop over the 9 conv taps; inner 8 cin-steps fully unrolled so all
// ds/global loads use immediate offsets. A-fragment addresses are branch-free:
// halo / M-padding rows select the always-zero LDS row via cndmask.
// sine_mode=1 computes the pos-branch sine embedding during the LDS fill,
// removing the pos_query HBM round-trip entirely.
// ---------------------------------------------------------------------------
__global__ __launch_bounds__(256) void conv_stack_kernel(
    const float* __restrict__ in,             // (B,49,256) fp32
    const unsigned short* __restrict__ wpack, // layers * 72*16*512 bf16
    const float* __restrict__ b0, const float* __restrict__ b1,
    const float* __restrict__ b2, const float* __restrict__ b3,
    int nlayers, int do_relu, int sine_mode,
    float* __restrict__ out)                  // (B,49,256) fp32
{
  __shared__ __align__(32) unsigned short act[2][ABUF];
  const int img  = blockIdx.x;
  const int tid  = threadIdx.x;
  const int lane = tid & 31, wave = tid >> 5;
  const int klane = lane >> 4, mrow = lane & 15;

  if (sine_mode) fill_sine(act[0], in + (size_t)img * FEAT, tid);
  else           fill_bf16(act[0], in + (size_t)img * FEAT, tid);
  if (tid < 256) act[1][FEAT + tid] = 0;
  __syncthreads();

  // per-lane M-tile geometry (constant across layers)
  int m_[4], oy_[4], ox_[4];
#pragma unroll
  for (int mt = 0; mt < 4; ++mt) {
    m_[mt]  = mt * 16 + mrow;
    oy_[mt] = m_[mt] / 7;
    ox_[mt] = m_[mt] - oy_[mt] * 7;
  }
  const int zidx = FEAT + (klane << 3);
  const int nt0 = wave * 2, nt1 = nt0 + 1;

  for (int l = 0; l < nlayers; ++l) {
    const unsigned short* cur = act[l & 1];
    unsigned short* nxt = act[(l + 1) & 1];
    const unsigned short* wl = wpack + (size_t)l * (72 * 16 * 512);
    const float* bias = (l == 0) ? b0 : (l == 1) ? b1 : (l == 2) ? b2 : b3;
    const unsigned short* wp0 = wl + ((nt0 * 32 + lane) << 4);
    const unsigned short* wp1 = wl + ((nt1 * 32 + lane) << 4);

    f32x8 acc[4][2];
#pragma unroll
    for (int mt = 0; mt < 4; ++mt) { acc[mt][0] = zero8(); acc[mt][1] = zero8(); }

    for (int dydx = 0; dydx < 9; ++dydx) {
      const int dy = dydx / 3 - 1;
      const int dx = dydx - (dydx / 3) * 3 - 1;
      const int delta = dy * 7 + dx;

      int aidx[4];
#pragma unroll
      for (int mt = 0; mt < 4; ++mt) {
        int iy = oy_[mt] + dy, ix = ox_[mt] + dx;
        bool valid = (m_[mt] < SEQ) & ((unsigned)iy < 7u) & ((unsigned)ix < 7u);
        aidx[mt] = valid ? ((m_[mt] + delta) * 256 + (klane << 3)) : zidx;
      }

      const unsigned short* wb0 = wp0 + (size_t)dydx * 65536;  // 8 steps * 8192 hw
      const unsigned short* wb1 = wp1 + (size_t)dydx * 65536;
      __builtin_prefetch(wb0 + 65536, 0, 3);   // next tap / next layer weights
      __builtin_prefetch(wb1 + 65536, 0, 3);

#pragma unroll
      for (int s = 0; s < 8; ++s) {
        CastBF bc0; bc0.u = *(const u16x16*)(wb0 + s * 8192);
        CastBF bc1; bc1.u = *(const u16x16*)(wb1 + s * 8192);
        bf16x16 af[4];
#pragma unroll
        for (int mt = 0; mt < 4; ++mt) {
          CastH8 av;
          av.h[0] = *(const u16x8*)(cur + aidx[mt] + s * 32);
          av.h[1] = *(const u16x8*)(cur + aidx[mt] + s * 32 + 16);
          CastBF ac; ac.u = av.v;
          af[mt] = ac.b;
        }
#pragma unroll
        for (int mt = 0; mt < 4; ++mt) {
          acc[mt][0] = wmma_bf16(af[mt], bc0.b, acc[mt][0]);
          acc[mt][1] = wmma_bf16(af[mt], bc1.b, acc[mt][1]);
        }
      }
    }
    __syncthreads();

    const int last = (l == nlayers - 1);
#pragma unroll
    for (int jj = 0; jj < 2; ++jj) {
      int n = (nt0 + jj) * 16 + mrow;
      float bv = bias[n];
#pragma unroll
      for (int mt = 0; mt < 4; ++mt) {
#pragma unroll
        for (int r = 0; r < 8; ++r) {
          int row = mt * 16 + klane * 8 + r;
          if (row < SEQ) {
            float v = acc[mt][jj][r] + bv;
            if (do_relu) v = fmaxf(v, 0.f);
            if (last) out[(size_t)img * FEAT + row * 256 + n] = v;
            else      nxt[row * 256 + n] = f2bf(v);
          }
        }
      }
    }
    __syncthreads();
  }
}

// ---------------------------------------------------------------------------
// 49-row WMMA GEMM helper: A (bf16, LDS buffer of ABUF entries incl zero row),
// packed W fragments. One N-tile x 4 M-tiles, K = 256 (8 steps, unrolled).
// ---------------------------------------------------------------------------
__device__ __forceinline__ void gemm49(const unsigned short* __restrict__ A,
                                       const unsigned short* __restrict__ Wp,
                                       int Ntiles, int nt, int lane,
                                       f32x8 acc[4]) {
  const int klane = lane >> 4, mrow = lane & 15;
  const unsigned short* wp = Wp + ((nt * 32 + lane) << 4);
  int aidx[4];
#pragma unroll
  for (int mt = 0; mt < 4; ++mt) {
    int m = mt * 16 + mrow;
    aidx[mt] = (m < SEQ ? m * 256 : FEAT) + (klane << 3);
  }
#pragma unroll
  for (int t = 0; t < 8; ++t) {
    CastBF bc; bc.u = *(const u16x16*)(wp + (size_t)t * (Ntiles * 512));
#pragma unroll
    for (int mt = 0; mt < 4; ++mt) {
      CastH8 av;
      av.h[0] = *(const u16x8*)(A + aidx[mt] + t * 32);
      av.h[1] = *(const u16x8*)(A + aidx[mt] + t * 32 + 16);
      CastBF ac; ac.u = av.v;
      acc[mt] = wmma_bf16(ac.b, bc.b, acc[mt]);
    }
  }
}

__device__ __forceinline__ void store_tiles_bf(unsigned short* dst, int stride,
                                               const f32x8 acc[4], int nt, int lane,
                                               const float* __restrict__ bias,
                                               int relu) {
  int n = nt * 16 + (lane & 15);
  int rbase = (lane >> 4) * 8;
  float bv = bias[n];
#pragma unroll
  for (int mt = 0; mt < 4; ++mt)
#pragma unroll
    for (int r = 0; r < 8; ++r) {
      int row = mt * 16 + rbase + r;
      if (row < SEQ) {
        float v = acc[mt][r] + bv;
        if (relu) v = fmaxf(v, 0.f);
        dst[row * stride + n] = f2bf(v);
      }
    }
}

// ---------------------------------------------------------------------------
// cls head: sigmoid(cls_features @ W(256x80) + b) -> all_proposals[:, :80]
// ---------------------------------------------------------------------------
__global__ __launch_bounds__(256) void cls_head_kernel(
    const float* __restrict__ feats, const unsigned short* __restrict__ wp,
    const float* __restrict__ bias, float* __restrict__ allp) {
  __shared__ __align__(32) unsigned short A[ABUF];
  const int img = blockIdx.x, tid = threadIdx.x;
  const int lane = tid & 31, wave = tid >> 5;
  fill_bf16(A, feats + (size_t)img * FEAT, tid);
  __syncthreads();
  if (wave < 5) {
    f32x8 acc[4];
#pragma unroll
    for (int mt = 0; mt < 4; ++mt) acc[mt] = zero8();
    gemm49(A, wp, 5, wave, lane, acc);
    int n = wave * 16 + (lane & 15);
    float bv = bias[n];
    int rbase = (lane >> 4) * 8;
#pragma unroll
    for (int mt = 0; mt < 4; ++mt)
#pragma unroll
      for (int r = 0; r < 8; ++r) {
        int row = mt * 16 + rbase + r;
        if (row < SEQ)
          allp[(size_t)img * SEQ * ALLP + row * ALLP + n] = sigmoidf_(acc[mt][r] + bv);
      }
  }
}

// ---------------------------------------------------------------------------
// repr_scores = max over 80 classes
// ---------------------------------------------------------------------------
__global__ void repr_kernel(const float* __restrict__ allp,
                            float* __restrict__ repr, int total) {
  int i = blockIdx.x * blockDim.x + threadIdx.x;
  if (i >= total) return;
  const float* p = allp + (size_t)i * ALLP;
  float m = p[0];
  for (int j = 1; j < CLS; ++j) m = fmaxf(m, p[j]);
  repr[i] = m;
}

// ---------------------------------------------------------------------------
// reg head (256->256->64->4) + pos head (256->256->2) + double-sigmoid bbox
// -> all_proposals[:, 80:84]
// ---------------------------------------------------------------------------
__global__ __launch_bounds__(256) void heads2_kernel(
    const float* __restrict__ regf, const float* __restrict__ posf,
    const unsigned short* __restrict__ rw0p, const float* __restrict__ rb0,
    const unsigned short* __restrict__ rw1p, const float* __restrict__ rb1,
    const float* __restrict__ rw2, const float* __restrict__ rb2,
    const unsigned short* __restrict__ qw0p, const float* __restrict__ qb0,
    const float* __restrict__ qw1, const float* __restrict__ qb1,
    float* __restrict__ allp) {
  __shared__ __align__(32) unsigned short A[ABUF];
  __shared__ __align__(32) unsigned short Bbuf[ABUF];
  __shared__ __align__(32) unsigned short H2[SEQ * 64];
  __shared__ float PC[SEQ * 2];
  const int img = blockIdx.x, tid = threadIdx.x;
  const int lane = tid & 31, wave = tid >> 5;

  // phase 1: reg_features -> A (bf16)
  fill_bf16(A, regf + (size_t)img * FEAT, tid);
  if (tid < 256) Bbuf[FEAT + tid] = 0;
  __syncthreads();

  // phase 2: h1 = relu(A @ reg_h_w0 + b0) -> Bbuf
  {
    f32x8 a0[4], a1[4];
#pragma unroll
    for (int mt = 0; mt < 4; ++mt) { a0[mt] = zero8(); a1[mt] = zero8(); }
    gemm49(A, rw0p, 16, wave * 2, lane, a0);
    gemm49(A, rw0p, 16, wave * 2 + 1, lane, a1);
    store_tiles_bf(Bbuf, 256, a0, wave * 2, lane, rb0, 1);
    store_tiles_bf(Bbuf, 256, a1, wave * 2 + 1, lane, rb0, 1);
  }
  __syncthreads();

  // phase 3: h2 = relu(h1 @ reg_h_w1 + b1) -> H2 (49 x 64)
  if (wave < 4) {
    f32x8 acc[4];
#pragma unroll
    for (int mt = 0; mt < 4; ++mt) acc[mt] = zero8();
    gemm49(Bbuf, rw1p, 4, wave, lane, acc);
    int n = wave * 16 + (lane & 15);
    float bv = rb1[n];
    int rbase = (lane >> 4) * 8;
#pragma unroll
    for (int mt = 0; mt < 4; ++mt)
#pragma unroll
      for (int r = 0; r < 8; ++r) {
        int row = mt * 16 + rbase + r;
        if (row < SEQ) H2[row * 64 + n] = f2bf(fmaxf(acc[mt][r] + bv, 0.f));
      }
  }
  __syncthreads();

  // phase 4: pos_features -> A
  fill_bf16(A, posf + (size_t)img * FEAT, tid);
  __syncthreads();

  // phase 5: p1 = relu(A @ pos_h_w0 + b0) -> Bbuf
  {
    f32x8 a0[4], a1[4];
#pragma unroll
    for (int mt = 0; mt < 4; ++mt) { a0[mt] = zero8(); a1[mt] = zero8(); }
    gemm49(A, qw0p, 16, wave * 2, lane, a0);
    gemm49(A, qw0p, 16, wave * 2 + 1, lane, a1);
    store_tiles_bf(Bbuf, 256, a0, wave * 2, lane, qb0, 1);
    store_tiles_bf(Bbuf, 256, a1, wave * 2 + 1, lane, qb0, 1);
  }
  __syncthreads();

  // phase 6: pos_center = p1 @ pos_h_w1 + b1  (49 x 2)
  if (tid < SEQ * 2) {
    int row = tid >> 1, c = tid & 1;
    float s = qb1[c];
    for (int k = 0; k < 256; ++k) s += bf2f(Bbuf[row * 256 + k]) * qw1[k * 2 + c];
    PC[row * 2 + c] = s;
  }
  __syncthreads();

  // phase 7: reg_out = sigmoid(h2 @ w2 + b2); bbox = sigmoid(reg_out + [pc,0,0])
  if (tid < SEQ * 4) {
    int row = tid >> 2, c = tid & 3;
    float s = rb2[c];
    for (int k = 0; k < 64; ++k) s += bf2f(H2[row * 64 + k]) * rw2[k * 4 + c];
    float ro = sigmoidf_(s);
    float off = (c < 2) ? PC[row * 2 + c] : 0.f;
    allp[(size_t)img * SEQ * ALLP + row * ALLP + CLS + c] = sigmoidf_(ro + off);
  }
}

// ---------------------------------------------------------------------------
// top-k selection (matches lax.top_k: descending, lowest index wins ties),
// gathers, and sine embedding of the top-k centers (d_model = 512, D = 4).
// ---------------------------------------------------------------------------
__global__ __launch_bounds__(256) void topk_kernel(
    const float* __restrict__ repr, const float* __restrict__ clsf,
    const float* __restrict__ regf, const float* __restrict__ allp,
    float* __restrict__ outProp, float* __restrict__ outCent,
    float* __restrict__ outPos) {
  __shared__ float sc[SEQ];
  __shared__ int sel[TOPK];
  __shared__ float cent[TOPK * 4];
  const int img = blockIdx.x, tid = threadIdx.x;
  if (tid < SEQ) sc[tid] = repr[img * SEQ + tid];
  __syncthreads();
  if (tid == 0) {
    for (int k = 0; k < TOPK; ++k) {
      float best = -3.4e38f; int bi = 0;
      for (int i = 0; i < SEQ; ++i)
        if (sc[i] > best) { best = sc[i]; bi = i; }
      sel[k] = bi; sc[bi] = -3.4e38f;
    }
  }
  __syncthreads();
  if (tid < TOPK * 4) {
    int k = tid >> 2, c = tid & 3;
    float v = allp[(size_t)img * SEQ * ALLP + sel[k] * ALLP + CLS + c];
    cent[tid] = v;
    outCent[(size_t)img * TOPK * 4 + tid] = v;
  }
  __syncthreads();
  // top_k_proposals = concat(cls_features, reg_features) gathered (32 x 512),
  // vectorized as float4 (b128) copies.
  for (int g = tid; g < TOPK * 128; g += 256) {
    int k = g >> 7, q = g & 127;
    int row = sel[k];
    const float* src = (q < 64)
        ? (clsf + (size_t)img * FEAT + row * 256 + q * 4)
        : (regf + (size_t)img * FEAT + row * 256 + (q - 64) * 4);
    *(f32x4*)(outProp + (size_t)img * TOPK * 512 + (size_t)g * 4) = *(const f32x4*)src;
  }
  // top_k_pos: blocks [emb(y), emb(x), emb(w), emb(h)], half = 256
  for (int e = tid; e < TOPK * 1024; e += 256) {
    int k = e >> 10, c = e & 1023;
    int blk = c >> 8, j = c & 255;
    int csel = (blk == 0) ? 1 : (blk == 1) ? 0 : blk;
    float coord = cent[k * 4 + csel];
    float freq = __builtin_exp2f((float)(j >> 1) * NEG_L2_1E4_D128);
    float arg = coord * TWO_PI * freq;
    outPos[(size_t)img * TOPK * 1024 + e] = (j & 1) ? __cosf(arg) : __sinf(arg);
  }
}

// ---------------------------------------------------------------------------
// Host-side launcher
// ---------------------------------------------------------------------------
extern "C" void kernel_launch(void* const* d_in, const int* in_sizes, int n_in,
                              void* d_out, int out_size, void* d_ws, size_t ws_size,
                              hipStream_t stream) {
  (void)in_sizes; (void)n_in; (void)out_size; (void)ws_size;
  const float* inputs = (const float*)d_in[0];
  const float* cw[4] = {(const float*)d_in[1], (const float*)d_in[3],
                        (const float*)d_in[5], (const float*)d_in[7]};
  const float* cb[4] = {(const float*)d_in[2], (const float*)d_in[4],
                        (const float*)d_in[6], (const float*)d_in[8]};
  const float* rw[4] = {(const float*)d_in[9], (const float*)d_in[11],
                        (const float*)d_in[13], (const float*)d_in[15]};
  const float* rb[4] = {(const float*)d_in[10], (const float*)d_in[12],
                        (const float*)d_in[14], (const float*)d_in[16]};
  const float* pw[3] = {(const float*)d_in[17], (const float*)d_in[19],
                        (const float*)d_in[21]};
  const float* pb[3] = {(const float*)d_in[18], (const float*)d_in[20],
                        (const float*)d_in[22]};
  const float* clsh_w = (const float*)d_in[23];
  const float* clsh_b = (const float*)d_in[24];
  const float* rh_w0 = (const float*)d_in[25]; const float* rh_b0 = (const float*)d_in[26];
  const float* rh_w1 = (const float*)d_in[27]; const float* rh_b1 = (const float*)d_in[28];
  const float* rh_w2 = (const float*)d_in[29]; const float* rh_b2 = (const float*)d_in[30];
  const float* ph_w0 = (const float*)d_in[31]; const float* ph_b0 = (const float*)d_in[32];
  const float* ph_w1 = (const float*)d_in[33]; const float* ph_b1 = (const float*)d_in[34];

  // Scratch layout (256-B aligned offsets)
  char* ws = (char*)d_ws;
  size_t off = 0;
  auto take = [&](size_t bytes) {
    char* p = ws + off; off += (bytes + 255) & ~(size_t)255; return p;
  };
  const size_t CONVPACK = 72ull * 16 * 512;                 // halfwords per layer
  unsigned short* packConv  = (unsigned short*)take(11ull * CONVPACK * 2);
  unsigned short* packClsH  = (unsigned short*)take(8ull * 5 * 512 * 2);
  unsigned short* packRegH0 = (unsigned short*)take(8ull * 16 * 512 * 2);
  unsigned short* packRegH1 = (unsigned short*)take(8ull * 4 * 512 * 2);
  unsigned short* packPosH0 = (unsigned short*)take(8ull * 16 * 512 * 2);
  float* clsf    = (float*)take((size_t)BATCH * FEAT * 4);
  float* regf    = (float*)take((size_t)BATCH * FEAT * 4);
  float* posfeat = (float*)take((size_t)BATCH * FEAT * 4);
  float* reprS   = (float*)take((size_t)BATCH * SEQ * 4);

  // --- pack weights into WMMA B layout ---
  const int convTotal = (int)CONVPACK;                      // 589824
  for (int i = 0; i < 4; ++i)
    packb_kernel<<<convTotal / 256, 256, 0, stream>>>(cw[i], packConv + (size_t)i * CONVPACK, 16, 256, convTotal);
  for (int i = 0; i < 4; ++i)
    packb_kernel<<<convTotal / 256, 256, 0, stream>>>(rw[i], packConv + (size_t)(4 + i) * CONVPACK, 16, 256, convTotal);
  for (int i = 0; i < 3; ++i)
    packb_kernel<<<convTotal / 256, 256, 0, stream>>>(pw[i], packConv + (size_t)(8 + i) * CONVPACK, 16, 256, convTotal);
  packb_kernel<<<80, 256, 0, stream>>>(clsh_w, packClsH, 5, 80, 8 * 5 * 512);
  packb_kernel<<<256, 256, 0, stream>>>(rh_w0, packRegH0, 16, 256, 8 * 16 * 512);
  packb_kernel<<<64, 256, 0, stream>>>(rh_w1, packRegH1, 4, 64, 8 * 4 * 512);
  packb_kernel<<<256, 256, 0, stream>>>(ph_w0, packPosH0, 16, 256, 8 * 16 * 512);

  // --- conv stacks (WMMA implicit GEMM); pos branch fuses the sine embed ---
  conv_stack_kernel<<<BATCH, 256, 0, stream>>>(inputs, packConv,
      cb[0], cb[1], cb[2], cb[3], 4, 0, 0, clsf);
  conv_stack_kernel<<<BATCH, 256, 0, stream>>>(inputs, packConv + 4ull * CONVPACK,
      rb[0], rb[1], rb[2], rb[3], 4, 0, 0, regf);
  conv_stack_kernel<<<BATCH, 256, 0, stream>>>(inputs, packConv + 8ull * CONVPACK,
      pb[0], pb[1], pb[2], pb[2], 3, 1, 1, posfeat);

  // --- outputs layout (floats) ---
  float* out = (float*)d_out;
  float* outProp = out;                                   // B*32*512
  float* outCent = out + 16777216;                        // B*32*4
  float* outPos  = out + 16908288;                        // B*32*1024
  float* outAll  = out + 50462720;                        // B*49*84

  cls_head_kernel<<<BATCH, 256, 0, stream>>>(clsf, packClsH, clsh_b, outAll);
  repr_kernel<<<(BATCH * SEQ + 255) / 256, 256, 0, stream>>>(outAll, reprS, BATCH * SEQ);
  heads2_kernel<<<BATCH, 256, 0, stream>>>(regf, posfeat,
      packRegH0, rh_b0, packRegH1, rh_b1, rh_w2, rh_b2,
      packPosH0, ph_b0, ph_w1, ph_b1, outAll);
  topk_kernel<<<BATCH, 256, 0, stream>>>(reprS, clsf, regf, outAll,
                                         outProp, outCent, outPos);
}
